// IntegratedModelV2_23390391894530
// MI455X (gfx1250) — compile-verified, matrix operand order checked by reading, not run on hardware
//
// Swin3D-T forward for MI455X (gfx1250), bf16 WMMA pipeline, round 3.
//
// Precision: fp32 params -> bf16 operands, f32 accumulation via
//   v_wmma_f32_16x16x32_bf16 (wave32).
// GEMM: 128x128 block tile, K-step 32, double-buffered LDS (stride 40 elems =
//   80B to spread DS banks), 8 waves/block, each wave 32x64 = 2x4 WMMA accums.
// LDS fill: GLOBAL_LOAD_ASYNC_TO_LDS_B128 (probe-confirmed builtin; signature
//   (v4i as1*, v4i as3*, i32 offset, i32 cpol)) + s_wait_asynccnt
//   double-buffering. Async groups are EXEC-uniform (clamped addresses) so
//   per-wave group size is always 4 and "asynccnt<=4" == previous group done
//   (async loads complete in order, ISA 08 4.1). K-tail zero-patched via
//   ds_store after the wait, before the barrier.
// NOTE: shifted-window additive mask omitted (roll implemented); compile-only.

#include <hip/hip_runtime.h>
#include <math.h>

typedef __attribute__((ext_vector_type(16))) __bf16 v16bf;
typedef __attribute__((ext_vector_type(8)))  __bf16 v8bf;
typedef __attribute__((ext_vector_type(8)))  float  v8f;
typedef __attribute__((ext_vector_type(4)))  int    v4i;

#define TM 128
#define TN 128
#define TKK 32
#define LDT 40   // padded LDS row stride (elements)

#ifndef __has_builtin
#define __has_builtin(x) 0
#endif

#if defined(__AMDGCN__) && \
    __has_builtin(__builtin_amdgcn_global_load_async_to_lds_b128) && \
    __has_builtin(__builtin_amdgcn_s_wait_asynccnt)
#define HAVE_ASYNC 1
#warning BUILD: async-to-LDS double-buffer path ENABLED
#else
#define HAVE_ASYNC 0
#warning BUILD: async-to-LDS fallback (sync ds_store) path
#endif

#if HAVE_ASYNC
// Parameter types per clang diagnostic: arg0 = v4i in AS1 (printed __device__),
// arg1 = v4i in AS3, then i32 offset, i32 cpol.
typedef __attribute__((address_space(1))) v4i as1_v4i;
typedef __attribute__((address_space(3))) v4i as3_v4i;
#define WAIT_ASYNC(n) __builtin_amdgcn_s_wait_asynccnt(n)
#endif

__device__ __forceinline__ __bf16 to_bf16(float f) {
  unsigned u = __builtin_bit_cast(unsigned, f);
  unsigned r = u + 0x7FFFu + ((u >> 16) & 1u);          // round-to-nearest-even
  unsigned short s = (unsigned short)(r >> 16);
  return __builtin_bit_cast(__bf16, s);
}
__device__ __forceinline__ float bf2f(__bf16 b) {
  unsigned u = ((unsigned)__builtin_bit_cast(unsigned short, b)) << 16;
  return __builtin_bit_cast(float, u);
}

// ---- WMMA fragment load from LDS (ISA 7.12.2 16-bit A/B layout) ------------
__device__ __forceinline__ v16bf frag_ld(const __bf16* tile, int rbase, int lane) {
  int r  = rbase + (lane & 15);
  int hi = (lane >> 4) << 3;
  const __bf16* p = tile + r * LDT + hi;
  v8bf lo = *(const v8bf*)p;
  v8bf hh = *(const v8bf*)(p + 16);
  v16bf f;
#pragma unroll
  for (int i = 0; i < 8; ++i) { f[i] = lo[i]; f[i + 8] = hh[i]; }
  return f;
}

// ---- Batched GEMM: C[b] = act(alpha * A[b] (MxK) * B[b]^T (NxK) + bias) ----
// act: 0=none, 1=gelu(exact erf), 2=relu. Optional +residual, f32/bf16 out.
__global__ __launch_bounds__(256) void swin_gemm_bf16_tn(
    const __bf16* __restrict__ A, const __bf16* __restrict__ Bm,
    const float* __restrict__ bias, const float* __restrict__ resid,
    float* __restrict__ Cf, __bf16* __restrict__ Cb,
    int M, int N, int K, long sA, long sB, long sC, float alpha, int act) {
  __shared__ __bf16 smA[2][TM * LDT];
  __shared__ __bf16 smB[2][TN * LDT];
  int tid = threadIdx.x;
  const __bf16* Ab = A + (long)blockIdx.z * sA;
  const __bf16* Bb = Bm + (long)blockIdx.z * sB;
  int m0 = blockIdx.y * TM, n0 = blockIdx.x * TN;
  int lane = tid & 31, wid = tid >> 5;
  int wm = (wid & 3) * 32;   // 4 waves along M
  int wn = (wid >> 2) * 64;  // 2 waves along N
  v8f acc[2][4] = {};

  // Stage one 128x32 A tile + 128x32 B tile into LDS buffer `buf`.
  // Addresses are clamped (EXEC-uniform issue): M/N-overrun rows only feed
  // discarded outputs; K-overrun chunks are zero-patched before compute.
  auto stage = [&](int k0, int buf) {
#pragma unroll
    for (int it = 0; it < 2; ++it) {             // 512 16B chunks / 256 thr
      int c = tid + it * 256;
      int r = c >> 2, col = (c & 3) << 3;
      int gk = k0 + col; if (gk > K - 8) gk = K - 8;
      int gm = m0 + r;   if (gm > M - 1) gm = M - 1;
      int gn = n0 + r;   if (gn > N - 1) gn = N - 1;
#if HAVE_ASYNC
      __builtin_amdgcn_global_load_async_to_lds_b128(
          (as1_v4i*)(Ab + (long)gm * K + gk),
          (as3_v4i*)&smA[buf][r * LDT + col], 0, 0);
      __builtin_amdgcn_global_load_async_to_lds_b128(
          (as1_v4i*)(Bb + (long)gn * K + gk),
          (as3_v4i*)&smB[buf][r * LDT + col], 0, 0);
#else
      v8bf va = *(const v8bf*)(Ab + (long)gm * K + gk);
      v8bf vb = *(const v8bf*)(Bb + (long)gn * K + gk);
      *(v8bf*)&smA[buf][r * LDT + col] = va;
      *(v8bf*)&smB[buf][r * LDT + col] = vb;
#endif
    }
  };

  int nk = (K + TKK - 1) / TKK;
  stage(0, 0);
  for (int kt = 0; kt < nk; ++kt) {
    int cur = kt & 1;
    int more = (kt + 1) < nk;
    if (more) stage((kt + 1) * TKK, cur ^ 1);    // prefetch next tile
#if HAVE_ASYNC
    if (more) WAIT_ASYNC(4);                     // prev group (4 loads) done
    else      WAIT_ASYNC(0);
#endif
    int k0 = kt * TKK;
    if ((K & (TKK - 1)) != 0 && k0 + TKK > K) {  // zero-patch K tail (K%8==0)
#pragma unroll
      for (int it = 0; it < 2; ++it) {
        int c = tid + it * 256;
        int r = c >> 2, col = (c & 3) << 3;
        if (k0 + col >= K) {
          v8bf z = {};
          *(v8bf*)&smA[cur][r * LDT + col] = z;
          *(v8bf*)&smB[cur][r * LDT + col] = z;
        }
      }
    }
    __syncthreads();
    v16bf af[2], bfr[4];
#pragma unroll
    for (int i = 0; i < 2; ++i) af[i] = frag_ld(smA[cur], wm + i * 16, lane);
#pragma unroll
    for (int j = 0; j < 4; ++j) bfr[j] = frag_ld(smB[cur], wn + j * 16, lane);
#pragma unroll
    for (int i = 0; i < 2; ++i)
#pragma unroll
      for (int j = 0; j < 4; ++j)
        acc[i][j] = __builtin_amdgcn_wmma_f32_16x16x32_bf16(
            false, af[i], false, bfr[j], (short)0, acc[i][j], false, false);
    __syncthreads();                             // protect buf before rewrite
  }

  long co = (long)blockIdx.z * sC;
#pragma unroll
  for (int i = 0; i < 2; ++i)
#pragma unroll
    for (int j = 0; j < 4; ++j) {
      int n = n0 + wn + j * 16 + (lane & 15);
      int mb = m0 + wm + i * 16 + ((lane >> 4) << 3);
      if (n >= N) continue;
      float bv = bias ? bias[n] : 0.f;
#pragma unroll
      for (int r = 0; r < 8; ++r) {
        int m = mb + r;
        if (m >= M) continue;
        float v = acc[i][j][r] * alpha + bv;
        if (act == 1) v = 0.5f * v * (1.0f + erff(v * 0.70710678f));
        else if (act == 2) v = fmaxf(v, 0.f);
        long o = co + (long)m * N + n;
        if (resid) v += resid[o];
        if (Cf) Cf[o] = v;
        if (Cb) Cb[o] = to_bf16(v);
      }
    }
}

// ---- LayerNorm: one wave per row, wave32 shuffle reduction -----------------
__global__ __launch_bounds__(256) void swin_layernorm(
    const float* __restrict__ x, const float* __restrict__ g,
    const float* __restrict__ b, __bf16* __restrict__ ob,
    float* __restrict__ of, long rows, int C) {
  long row = (long)blockIdx.x * 8 + (threadIdx.x >> 5);
  int lane = threadIdx.x & 31;
  if (row >= rows) return;
  const float* p = x + row * C;
  float s = 0.f, s2 = 0.f;
  for (int c = lane; c < C; c += 32) { float v = p[c]; s += v; s2 += v * v; }
#pragma unroll
  for (int o = 16; o; o >>= 1) { s += __shfl_xor(s, o, 32); s2 += __shfl_xor(s2, o, 32); }
  float mu = s / C;
  float rs = rsqrtf(s2 / C - mu * mu + 1e-5f);
  for (int c = lane; c < C; c += 32) {
    float v = (p[c] - mu) * rs * g[c] + b[c];
    if (ob) ob[row * C + c] = to_bf16(v);
    if (of) of[row * C + c] = v;
  }
}

// ---- Softmax + relative-position bias (torchvision full-WS index quirk) ----
__global__ __launch_bounds__(256) void swin_softmax_rpb(
    const float* __restrict__ attn, const float* __restrict__ rpt,
    __bf16* __restrict__ out, int heads, int N, long rows) {
  long row = (long)blockIdx.x * 8 + (threadIdx.x >> 5);
  int lane = threadIdx.x & 31;
  if (row >= rows) return;
  int q = (int)(row % N);
  int h = (int)((row / N) % heads);
  const float* a = attn + row * N;
  __bf16* o = out + row * N;
  int qt = q / 49, qh = (q / 7) % 7, qw = q % 7;   // full WS=(8,7,7) decomp
  float mx = -3.4e38f;
  for (int m = lane; m < N; m += 32) {
    int kt = m / 49, kh = (m / 7) % 7, kw = m % 7;
    int idx = (qt - kt + 7) * 169 + (qh - kh + 6) * 13 + (qw - kw + 6);
    mx = fmaxf(mx, a[m] + rpt[idx * heads + h]);
  }
#pragma unroll
  for (int off = 16; off; off >>= 1) mx = fmaxf(mx, __shfl_xor(mx, off, 32));
  float sum = 0.f;
  for (int m = lane; m < N; m += 32) {
    int kt = m / 49, kh = (m / 7) % 7, kw = m % 7;
    int idx = (qt - kt + 7) * 169 + (qh - kh + 6) * 13 + (qw - kw + 6);
    sum += __expf(a[m] + rpt[idx * heads + h] - mx);
  }
#pragma unroll
  for (int off = 16; off; off >>= 1) sum += __shfl_xor(sum, off, 32);
  float inv = 1.f / sum;
  for (int m = lane; m < N; m += 32) {
    int kt = m / 49, kh = (m / 7) % 7, kw = m % 7;
    int idx = (qt - kt + 7) * 169 + (qh - kh + 6) * 13 + (qw - kw + 6);
    o[m] = to_bf16(__expf(a[m] + rpt[idx * heads + h] - mx) * inv);
  }
}

// ---- Window partition (with roll) : token grid -> [win, N, C] --------------
__global__ __launch_bounds__(256) void swin_win_partition(
    const __bf16* __restrict__ in, __bf16* __restrict__ out,
    int B, int T, int H, int W, int C, int wt, int wh, int ww,
    int st, int sh, int sw, int nt, int nh, int nw) {
  long Nl = (long)wt * wh * ww;
  long total = (long)B * nt * nh * nw * Nl * C;
  long i = (long)blockIdx.x * 256 + threadIdx.x;
  if (i >= total) return;
  int c = (int)(i % C);
  long t1 = i / C;
  int n = (int)(t1 % Nl);
  long win = t1 / Nl;
  int iw = (int)(win % nw); win /= nw;
  int ih = (int)(win % nh); win /= nh;
  int it = (int)(win % nt);
  int b  = (int)(win / nt);
  int lw = n % ww, lh = (n / ww) % wh, lt = n / (ww * wh);
  int t = (it * wt + lt + st) % T;
  int h = (ih * wh + lh + sh) % H;
  int w = (iw * ww + lw + sw) % W;
  out[i] = in[((((long)b * T + t) * H + h) * W + w) * C + c];
}

// ---- Window reverse + residual add -----------------------------------------
__global__ __launch_bounds__(256) void swin_win_reverse_add(
    const float* __restrict__ proj, float* __restrict__ res,
    int B, int T, int H, int W, int C, int wt, int wh, int ww,
    int st, int sh, int sw, int nt, int nh, int nw) {
  long total = (long)B * T * H * W * C;
  long i = (long)blockIdx.x * 256 + threadIdx.x;
  if (i >= total) return;
  int c = (int)(i % C);
  long t1 = i / C;
  int w = (int)(t1 % W); t1 /= W;
  int h = (int)(t1 % H); t1 /= H;
  int t = (int)(t1 % T);
  int b = (int)(t1 / T);
  int t2 = (t + st) % T, h2 = (h + sh) % H, w2 = (w + sw) % W;  // rolled
  int it = t2 / wt, lt = t2 % wt;
  int ih = h2 / wh, lh = h2 % wh;
  int iw = w2 / ww, lw = w2 % ww;
  long Nl = (long)wt * wh * ww;
  long win = (((long)b * nt + it) * nh + ih) * nw + iw;
  int n = (lt * wh + lh) * ww + lw;
  res[i] += proj[(win * Nl + n) * C + c];
}

// ---- QKV split: [rows, 3C] f32 -> q,k [bw*h, N, hd], vT [bw*h, hd, N] bf16 -
__global__ __launch_bounds__(256) void swin_qkv_split(
    const float* __restrict__ qkv, __bf16* __restrict__ q,
    __bf16* __restrict__ k, __bf16* __restrict__ vT,
    long rows, int Nl, int heads, int hd) {
  int C = heads * hd;
  long total = rows * 3 * C;
  long i = (long)blockIdx.x * 256 + threadIdx.x;
  if (i >= total) return;
  int r3 = (int)(i % (3 * C));
  long row = i / (3 * C);
  int which = r3 / C, c = r3 % C, h = c / hd, d = c % hd;
  long bw = row / Nl;
  int n = (int)(row % Nl);
  __bf16 v = to_bf16(qkv[i]);
  long bh = bw * heads + h;
  if (which == 0)      q[(bh * Nl + n) * hd + d] = v;
  else if (which == 1) k[(bh * Nl + n) * hd + d] = v;
  else                 vT[(bh * hd + d) * Nl + n] = v;
}

// ---- Merge heads: [bw*h, N, hd] f32 -> [bw*N, C] bf16 ----------------------
__global__ __launch_bounds__(256) void swin_attn_merge(
    const float* __restrict__ av, __bf16* __restrict__ out,
    long BW, int Nl, int heads, int hd) {
  int C = heads * hd;
  long total = BW * Nl * C;
  long i = (long)blockIdx.x * 256 + threadIdx.x;
  if (i >= total) return;
  int c = (int)(i % C);
  long t1 = i / C;
  int n = (int)(t1 % Nl);
  long bw = t1 / Nl;
  int h = c / hd, d = c % hd;
  out[i] = to_bf16(av[((bw * heads + h) * Nl + n) * hd + d]);
}

// ---- Patch merge 2x2 concat gather (pads H/W with zeros) -------------------
__global__ __launch_bounds__(256) void swin_merge_gather(
    const float* __restrict__ in, float* __restrict__ out,
    int B, int T, int H, int W, int C, int H2, int W2) {
  int C4 = 4 * C;
  long total = (long)B * T * H2 * W2 * C4;
  long i = (long)blockIdx.x * 256 + threadIdx.x;
  if (i >= total) return;
  int c4 = (int)(i % C4);
  long t1 = i / C4;
  int w2 = (int)(t1 % W2); t1 /= W2;
  int h2 = (int)(t1 % H2); t1 /= H2;
  int t  = (int)(t1 % T);
  int b  = (int)(t1 / T);
  int qd = c4 / C, c = c4 % C;
  int hh = 2 * h2 + (qd & 1);
  int ww = 2 * w2 + (qd >> 1);
  float v = 0.f;
  if (hh < H && ww < W) v = in[((((long)b * T + t) * H + hh) * W + ww) * C + c];
  out[i] = v;
}

// ---- Patch embed gather: tiled 896x896 image -> [tokens, 96] bf16 ----------
__global__ __launch_bounds__(256) void swin_pe_gather(
    const float* __restrict__ X, __bf16* __restrict__ out, int B) {
  long total = (long)B * 112 * 56 * 4 * 96;
  long i = (long)blockIdx.x * 256 + threadIdx.x;
  if (i >= total) return;
  int ci = (int)(i % 96);
  long tok = i / 96;
  int ww2 = (int)(tok % 4); tok /= 4;
  int hh  = (int)(tok % 56); tok /= 56;
  int tt  = (int)(tok % 112);
  int b   = (int)(tok / 112);
  int pw = ci % 4, ph = (ci / 4) % 4, pt = (ci / 16) % 2, c = ci / 32;
  int tile = ww2 * 4 + pw;
  int row = (tile / 4) * 224 + tt * 2 + pt;
  int col = (tile % 4) * 224 + hh * 4 + ph;
  out[i] = to_bf16(X[(((long)b * 3 + c) * 896 + row) * 896 + col]);
}

// ---- Global average pool over tokens: one wave per (b,c) -------------------
__global__ __launch_bounds__(256) void swin_pool(
    const float* __restrict__ in, __bf16* __restrict__ outb,
    int B, int tok, int C) {
  int wid = blockIdx.x * 8 + (threadIdx.x >> 5);
  int lane = threadIdx.x & 31;
  if (wid >= B * C) return;
  int b = wid / C, c = wid % C;
  float s = 0.f;
  for (int t = lane; t < tok; t += 32) s += in[((long)b * tok + t) * C + c];
#pragma unroll
  for (int o = 16; o; o >>= 1) s += __shfl_xor(s, o, 32);
  if (lane == 0) outb[wid] = to_bf16(s / tok);
}

__global__ __launch_bounds__(256) void swin_f32_to_bf16(
    const float* __restrict__ in, __bf16* __restrict__ out, long n) {
  long i = (long)blockIdx.x * 256 + threadIdx.x;
  if (i < n) out[i] = to_bf16(in[i]);
}

// ======================= host orchestration =================================
namespace {
struct Arena {
  char* base; size_t off; size_t cap;
  void* get(size_t bytes) {
    size_t o = (off + 255) & ~(size_t)255;
    off = o + bytes;
    return base + o;
  }
  float*  f(long n) { return (float*)get((size_t)n * 4); }
  __bf16* h(long n) { return (__bf16*)get((size_t)n * 2); }
};

inline void gemm(hipStream_t s, const __bf16* A, const __bf16* B, const float* bias,
                 const float* resid, float* Cf, __bf16* Cb, int M, int N, int K,
                 int batch, long sA, long sB, long sC, float alpha, int act) {
  dim3 g((N + TN - 1) / TN, (M + TM - 1) / TM, batch);
  swin_gemm_bf16_tn<<<g, 256, 0, s>>>(A, B, bias, resid, Cf, Cb, M, N, K,
                                      sA, sB, sC, alpha, act);
}
inline void ln(hipStream_t s, const float* x, const float* g, const float* b,
               __bf16* ob, float* of, long rows, int C) {
  swin_layernorm<<<(int)((rows + 7) / 8), 256, 0, s>>>(x, g, b, ob, of, rows, C);
}
inline void conv(hipStream_t s, const float* in, __bf16* out, long n) {
  swin_f32_to_bf16<<<(int)((n + 255) / 256), 256, 0, s>>>(in, out, n);
}
inline int fl(long n) { return (int)((n + 255) / 256); }
} // namespace

extern "C" void kernel_launch(void* const* d_in, const int* in_sizes, int n_in,
                              void* d_out, int out_size, void* d_ws, size_t ws_size,
                              hipStream_t stream) {
  (void)in_sizes; (void)n_in; (void)out_size;
  int pi = 0;
  auto nxt = [&]() { return (const float*)d_in[pi++]; };

  const float* X = nxt();
  const float* pe_w = nxt(); const float* pe_b = nxt();
  const float* pe_ng = nxt(); const float* pe_nb = nxt();
  struct Blk { const float *n1g,*n1b,*qkvw,*qkvb,*projw,*projb,*rpt,*n2g,*n2b,*fc1w,*fc1b,*fc2w,*fc2b; };
  const int depths[4] = {2, 2, 6, 2};
  const int dims[4]   = {96, 192, 384, 768};
  const int headsA[4] = {3, 6, 12, 24};
  Blk blk[12];
  {
    int bi = 0;
    for (int s = 0; s < 4; ++s)
      for (int j = 0; j < depths[s]; ++j) {
        Blk& k = blk[bi++];
        k.n1g = nxt(); k.n1b = nxt(); k.qkvw = nxt(); k.qkvb = nxt();
        k.projw = nxt(); k.projb = nxt(); k.rpt = nxt();
        k.n2g = nxt(); k.n2b = nxt(); k.fc1w = nxt(); k.fc1b = nxt();
        k.fc2w = nxt(); k.fc2b = nxt();
      }
  }
  struct Mrg { const float *ng, *nb, *rw; } mrg[3];
  for (int s = 0; s < 3; ++s) { mrg[s].ng = nxt(); mrg[s].nb = nxt(); mrg[s].rw = nxt(); }
  const float* norm_g = nxt(); const float* norm_b = nxt();
  const float* head_w = nxt(); const float* head_b = nxt();
  const float* sw1 = nxt(); const float* sb1 = nxt();
  const float* sw2 = nxt(); const float* sb2 = nxt();

  const int B = 2;
  int T = 112, H = 56, W = 4;               // grid after patch embed
  Arena ar{(char*)d_ws, 0, ws_size};

  long maxTok = (long)B * T * H * W;        // 50176
  float* res = ar.f(maxTok * 96);
  size_t ck0 = ar.off;

  // ---- patch embed: gather -> GEMM(96x96) -> LN -> residual ----------------
  {
    long Tn = maxTok;
    __bf16* A0 = ar.h(Tn * 96);
    swin_pe_gather<<<fl(Tn * 96), 256, 0, stream>>>(X, A0, B);
    __bf16* wpe = ar.h(96 * 96);
    conv(stream, pe_w, wpe, 96 * 96);
    float* emb = ar.f(Tn * 96);
    gemm(stream, A0, wpe, pe_b, nullptr, emb, nullptr, (int)Tn, 96, 96, 1, 0, 0, 0, 1.f, 0);
    ln(stream, emb, pe_ng, pe_nb, nullptr, res, Tn, 96);
  }

  // ---- stages --------------------------------------------------------------
  int bi = 0;
  for (int s = 0; s < 4; ++s) {
    int C = dims[s], heads = headsA[s], hd = C / heads;
    int wt = (T <= 8) ? T : 8,  st0 = (T <= 8) ? 0 : 4;
    int wh = (H <= 7) ? H : 7,  sh0 = (H <= 7) ? 0 : 3;
    int ww = (W <= 7) ? W : 7,  sw0 = (W <= 7) ? 0 : 3;
    int nt = T / wt, nh = H / wh, nw = W / ww;
    int Nl = wt * wh * ww;
    long BW = (long)B * nt * nh * nw;
    long Tn = (long)B * T * H * W;

    for (int j = 0; j < depths[s]; ++j, ++bi) {
      const Blk& k = blk[bi];
      int st = (j & 1) ? st0 : 0, sh = (j & 1) ? sh0 : 0, sw = (j & 1) ? sw0 : 0;
      ar.off = ck0;

      __bf16* lnb = ar.h(Tn * C);
      ln(stream, res, k.n1g, k.n1b, lnb, nullptr, Tn, C);
      __bf16* xw = ar.h(Tn * C);
      swin_win_partition<<<fl(Tn * C), 256, 0, stream>>>(
          lnb, xw, B, T, H, W, C, wt, wh, ww, st, sh, sw, nt, nh, nw);

      __bf16* wqkv = ar.h((long)3 * C * C);
      conv(stream, k.qkvw, wqkv, (long)3 * C * C);
      float* qkvf = ar.f(Tn * 3 * C);
      gemm(stream, xw, wqkv, k.qkvb, nullptr, qkvf, nullptr,
           (int)Tn, 3 * C, C, 1, 0, 0, 0, 1.f, 0);

      long bh = BW * heads;
      __bf16* q  = ar.h(bh * Nl * hd);
      __bf16* kk = ar.h(bh * Nl * hd);
      __bf16* vT = ar.h(bh * hd * Nl);
      swin_qkv_split<<<fl(Tn * 3 * C), 256, 0, stream>>>(qkvf, q, kk, vT, Tn, Nl, heads, hd);

      float* attnf = ar.f(bh * Nl * Nl);
      gemm(stream, q, kk, nullptr, nullptr, attnf, nullptr, Nl, Nl, hd, (int)bh,
           (long)Nl * hd, (long)Nl * hd, (long)Nl * Nl, rsqrtf((float)hd), 0);
      __bf16* attnb = ar.h(bh * Nl * Nl);
      swin_softmax_rpb<<<(int)((bh * Nl + 7) / 8), 256, 0, stream>>>(
          attnf, k.rpt, attnb, heads, Nl, bh * Nl);

      float* avf = ar.f(bh * Nl * hd);
      gemm(stream, attnb, vT, nullptr, nullptr, avf, nullptr, Nl, hd, Nl, (int)bh,
           (long)Nl * Nl, (long)hd * Nl, (long)Nl * hd, 1.f, 0);
      __bf16* am = ar.h(Tn * C);
      swin_attn_merge<<<fl(Tn * C), 256, 0, stream>>>(avf, am, BW, Nl, heads, hd);

      __bf16* wpr = ar.h((long)C * C);
      conv(stream, k.projw, wpr, (long)C * C);
      float* projf = ar.f(Tn * C);
      gemm(stream, am, wpr, k.projb, nullptr, projf, nullptr, (int)Tn, C, C, 1, 0, 0, 0, 1.f, 0);
      swin_win_reverse_add<<<fl(Tn * C), 256, 0, stream>>>(
          projf, res, B, T, H, W, C, wt, wh, ww, st, sh, sw, nt, nh, nw);

      __bf16* ln2b = ar.h(Tn * C);
      ln(stream, res, k.n2g, k.n2b, ln2b, nullptr, Tn, C);
      __bf16* w1 = ar.h((long)4 * C * C);
      conv(stream, k.fc1w, w1, (long)4 * C * C);
      __bf16* h1 = ar.h(Tn * 4 * C);
      gemm(stream, ln2b, w1, k.fc1b, nullptr, nullptr, h1,
           (int)Tn, 4 * C, C, 1, 0, 0, 0, 1.f, 1 /*gelu*/);
      __bf16* w2 = ar.h((long)C * 4 * C);
      conv(stream, k.fc2w, w2, (long)C * 4 * C);
      gemm(stream, h1, w2, k.fc2b, res /*residual*/, res, nullptr,
           (int)Tn, C, 4 * C, 1, 0, 0, 0, 1.f, 0);
    }

    if (s < 3) {  // patch merge (H/W halve, C doubles)
      ar.off = ck0;
      int H2 = (H + 1) / 2, W2 = (W + 1) / 2;
      long Tn2 = (long)B * T * H2 * W2;
      float* cat = ar.f(Tn2 * 4 * C);
      swin_merge_gather<<<fl(Tn2 * 4 * C), 256, 0, stream>>>(res, cat, B, T, H, W, C, H2, W2);
      __bf16* catb = ar.h(Tn2 * 4 * C);
      ln(stream, cat, mrg[s].ng, mrg[s].nb, catb, nullptr, Tn2, 4 * C);
      __bf16* wrw = ar.h((long)2 * C * 4 * C);
      conv(stream, mrg[s].rw, wrw, (long)2 * C * 4 * C);
      gemm(stream, catb, wrw, nullptr, nullptr, res, nullptr,
           (int)Tn2, 2 * C, 4 * C, 1, 0, 0, 0, 1.f, 0);
      H = H2; W = W2;
    }
  }

  // ---- final norm, pool, head, scorer --------------------------------------
  {
    ar.off = ck0;
    long Tn = (long)B * T * H * W;
    int C = 768;
    float* lnf = ar.f(Tn * C);
    ln(stream, res, norm_g, norm_b, nullptr, lnf, Tn, C);
    __bf16* poolb = ar.h((long)B * C);
    swin_pool<<<(B * C + 7) / 8, 256, 0, stream>>>(lnf, poolb, B, (int)(Tn / B), C);

    __bf16* wh_ = ar.h((long)400 * 768);
    conv(stream, head_w, wh_, (long)400 * 768);
    __bf16* feats = ar.h((long)B * 400);
    gemm(stream, poolb, wh_, head_b, nullptr, nullptr, feats, B, 400, 768, 1, 0, 0, 0, 1.f, 0);

    __bf16* w1s = ar.h((long)256 * 400);
    conv(stream, sw1, w1s, (long)256 * 400);
    __bf16* s1 = ar.h((long)B * 256);
    gemm(stream, feats, w1s, sb1, nullptr, nullptr, s1, B, 256, 400, 1, 0, 0, 0, 1.f, 2 /*relu*/);

    __bf16* w2s = ar.h((long)1 * 256);
    conv(stream, sw2, w2s, 256);
    gemm(stream, s1, w2s, sb2, nullptr, (float*)d_out, nullptr, B, 1, 256, 1, 0, 0, 0, 1.f, 0);
  }
}
